// DualEncoder_70858370450221
// MI455X (gfx1250) — matrix operand order repaired and checked by reading, not compiled, not run on hardware
//
#include <hip/hip_runtime.h>
#include <hip/hip_bf16.h>
#include <cstdint>
#include <cstddef>

typedef __attribute__((ext_vector_type(16))) _Float16 v16h;
typedef __attribute__((ext_vector_type(8)))  _Float16 v8h;
typedef __attribute__((ext_vector_type(8)))  float    v8f;

#define FBIG 3.402823466e+38f

// ------------------------------------------------------------------ fill
__global__ void k_fill(float* __restrict__ p, float v, size_t n) {
  size_t i = (size_t)blockIdx.x * blockDim.x + threadIdx.x;
  if (i < n) p[i] = v;
}

// ------------------------------------------------------------------ WMMA GEMM
// C[M,N] = A[M,K] @ W[K,N] (+bias)(+ReLU). fp32 I/O, f16 WMMA, f32 accum.
// Block: 256 threads = 8 waves arranged 2(M)x4(N); each wave owns a 32x32
// C macro-tile = 2x2 accumulators -> 4 WMMAs per 32-K step.
// Block tile: 64(M) x 128(N), K stepped by 32.
__global__ __launch_bounds__(256) void k_gemm(const float* __restrict__ A,
                                              const float* __restrict__ W,
                                              const float* __restrict__ bias,
                                              float* __restrict__ C,
                                              int M, int K, int N, int relu) {
  // stride 40 halves (80B, multiple of 16B): b128-aligned rows, no bank pathologies
  __shared__ __align__(16) _Float16 As[64 * 40];   // 64 rows x 32 k
  __shared__ __align__(16) _Float16 Bs[128 * 40];  // 128 cols x 32 k (transposed)

  const int tid = threadIdx.x;
  const int rowBase = blockIdx.y * 64;
  const int colBase = blockIdx.x * 128;
  const int lane = tid & 31;
  const int w   = tid >> 5;    // wave 0..7
  const int wMr = w >> 2;      // 0..1 -> M offset wMr*32
  const int wNc = w & 3;       // 0..3 -> N offset wNc*32
  const int mL  = lane & 15;   // row (A) / col (B,C) within 16
  const int gL  = lane >> 4;   // half-wave group

  v8f acc00 = {}, acc01 = {}, acc10 = {}, acc11 = {};

  for (int kBase = 0; kBase < K; kBase += 32) {
    // ---- stage A tile: 64 rows x 32 k (2 float4 per thread)
#pragma unroll
    for (int t = 0; t < 2; ++t) {
      int lin = tid + t * 256;
      int r   = lin >> 3;        // 0..63
      int k4  = (lin & 7) * 4;   // 0..28
      int grow = rowBase + r;
      float4 av = make_float4(0.f, 0.f, 0.f, 0.f);
      if (grow < M) {
        av = *(const float4*)(A + (size_t)grow * K + kBase + k4);
        if (kBase + 32 < K)
          __builtin_prefetch(A + (size_t)grow * K + kBase + 32 + k4, 0, 1);
      }
      _Float16* d = &As[r * 40 + k4];
      d[0] = (_Float16)av.x; d[1] = (_Float16)av.y;
      d[2] = (_Float16)av.z; d[3] = (_Float16)av.w;
    }
    // ---- stage B tile transposed: Bs[n][k], 128 cols x 32 k (4 float4 per thread)
#pragma unroll
    for (int t = 0; t < 4; ++t) {
      int lin = tid + t * 256;       // 0..1023
      int kk  = lin >> 5;            // 0..31
      int n4  = (lin & 31) * 4;      // 0..124
      float4 bv = make_float4(0.f, 0.f, 0.f, 0.f);
      if (colBase + n4 < N)
        bv = *(const float4*)(W + (size_t)(kBase + kk) * N + colBase + n4);
      Bs[(n4 + 0) * 40 + kk] = (_Float16)bv.x;
      Bs[(n4 + 1) * 40 + kk] = (_Float16)bv.y;
      Bs[(n4 + 2) * 40 + kk] = (_Float16)bv.z;
      Bs[(n4 + 3) * 40 + kk] = (_Float16)bv.w;
    }
    __syncthreads();

    // ---- fragments per ISA 16-bit layouts (A 16x32 MxK, B 32x16 KxN)
    v16h a0, a1, b0, b1;
    {
      const int r0 = (wMr * 32 + mL) * 40;        // A rows 0..15 of macro-tile
      const int r1 = (wMr * 32 + 16 + mL) * 40;   // A rows 16..31
      v8h lo, hi;
      lo = *(const v8h*)(&As[r0 + gL * 8]);
      hi = *(const v8h*)(&As[r0 + 16 + gL * 8]);
#pragma unroll
      for (int i = 0; i < 8; ++i) { a0[i] = lo[i]; a0[i + 8] = hi[i]; }
      lo = *(const v8h*)(&As[r1 + gL * 8]);
      hi = *(const v8h*)(&As[r1 + 16 + gL * 8]);
#pragma unroll
      for (int i = 0; i < 8; ++i) { a1[i] = lo[i]; a1[i + 8] = hi[i]; }
      const int c0 = (wNc * 32 + mL) * 40;        // B cols 0..15 of macro-tile
      const int c1 = (wNc * 32 + 16 + mL) * 40;   // B cols 16..31
      lo = *(const v8h*)(&Bs[c0 + gL * 8]);
      hi = *(const v8h*)(&Bs[c0 + 16 + gL * 8]);
#pragma unroll
      for (int i = 0; i < 8; ++i) { b0[i] = lo[i]; b0[i + 8] = hi[i]; }
      lo = *(const v8h*)(&Bs[c1 + gL * 8]);
      hi = *(const v8h*)(&Bs[c1 + 16 + gL * 8]);
#pragma unroll
      for (int i = 0; i < 8; ++i) { b1[i] = lo[i]; b1[i + 8] = hi[i]; }
    }
    acc00 = __builtin_amdgcn_wmma_f32_16x16x32_f16(false, a0, false, b0, (short)0, acc00, false, false);
    acc01 = __builtin_amdgcn_wmma_f32_16x16x32_f16(false, a0, false, b1, (short)0, acc01, false, false);
    acc10 = __builtin_amdgcn_wmma_f32_16x16x32_f16(false, a1, false, b0, (short)0, acc10, false, false);
    acc11 = __builtin_amdgcn_wmma_f32_16x16x32_f16(false, a1, false, b1, (short)0, acc11, false, false);
    __syncthreads();
  }

  // ---- store C (D layout: lane -> col = L%16; vgpr r -> row = r + 8*(L/16))
#pragma unroll
  for (int mf = 0; mf < 2; ++mf) {
#pragma unroll
    for (int nf = 0; nf < 2; ++nf) {
      const v8f* accp = (mf == 0) ? (nf == 0 ? &acc00 : &acc01)
                                  : (nf == 0 ? &acc10 : &acc11);
      int col = colBase + wNc * 32 + nf * 16 + mL;
      if (col >= N) continue;
      float bv = bias ? bias[col] : 0.f;
#pragma unroll
      for (int r = 0; r < 8; ++r) {
        int row = rowBase + wMr * 32 + mf * 16 + r + 8 * gL;
        if (row < M) {
          float v = (*accp)[r] + bv;
          if (relu) v = fmaxf(v, 0.f);
          C[(size_t)row * N + col] = v;
        }
      }
    }
  }
}

// ------------------------------------------------------------------ degree / norm
__global__ void k_deg(const int* __restrict__ dst, float* __restrict__ deg, int E) {
  int e = blockIdx.x * blockDim.x + threadIdx.x;
  if (e < E) atomicAdd(&deg[dst[e]], 1.0f);
}
__global__ void k_rsqrt_inplace(float* __restrict__ d, int n) {
  int i = blockIdx.x * blockDim.x + threadIdx.x;
  if (i < n) d[i] = rsqrtf(d[i]);
}

// ------------------------------------------------------------------ edge scatter
// agg[dst] += h[src] * dinv[src]*dinv[dst]   (float4 per thread)
__global__ void k_scatter(const int* __restrict__ src, const int* __restrict__ dst,
                          const float* __restrict__ dinv, const float* __restrict__ h,
                          float* __restrict__ agg, long long total, int F4, int F) {
  long long idx = (long long)blockIdx.x * blockDim.x + threadIdx.x;
  if (idx >= total) return;
  int e = (int)(idx / F4);
  int c = (int)(idx % F4) * 4;
  int s = src[e], d = dst[e];
  float nrm = dinv[s] * dinv[d];
  float4 hv = *(const float4*)(h + (size_t)s * F + c);
  float* out = agg + (size_t)d * F + c;
  atomicAdd(out + 0, hv.x * nrm);
  atomicAdd(out + 1, hv.y * nrm);
  atomicAdd(out + 2, hv.z * nrm);
  atomicAdd(out + 3, hv.w * nrm);
}

// ------------------------------------------------------------------ conv finalize
// out = agg + h * dinv^2 + bias  (+ReLU)
__global__ void k_finalize(const float* __restrict__ agg, const float* __restrict__ h,
                           const float* __restrict__ dinv, const float* __restrict__ bias,
                           float* __restrict__ out, int Nn, int F, int relu) {
  size_t idx = (size_t)blockIdx.x * blockDim.x + threadIdx.x;
  if (idx >= (size_t)Nn * F) return;
  int i = (int)(idx / F);
  int f = (int)(idx % F);
  float di = dinv[i];
  float v = agg[idx] + h[idx] * di * di + bias[f];
  if (relu) v = fmaxf(v, 0.f);
  out[idx] = v;
}

// ------------------------------------------------------------------ pooling
__global__ void k_count(const int* __restrict__ batch, float* __restrict__ cnt, int n) {
  int i = blockIdx.x * blockDim.x + threadIdx.x;
  if (i < n) atomicAdd(&cnt[batch[i]], 1.0f);
}
__global__ void k_pool_init(float* __restrict__ zg) {  // [64, 512]: mean | max
  int idx = blockIdx.x * blockDim.x + threadIdx.x;
  if (idx >= 64 * 256) return;
  int b = idx >> 8, f = idx & 255;
  zg[b * 512 + f] = 0.f;
  zg[b * 512 + 256 + f] = -FBIG;
}
__device__ __forceinline__ void atomicMaxF(float* addr, float val) {
  unsigned int* a = (unsigned int*)addr;
  unsigned int old = *a;
  while (true) {
    float f = __uint_as_float(old);
    if (f >= val) break;
    unsigned int assumed = old;
    old = atomicCAS(a, assumed, __float_as_uint(val));
    if (old == assumed) break;
  }
}
__global__ void k_pool(const float* __restrict__ z, const int* __restrict__ batch,
                       float* __restrict__ zg, int Nn) {
  size_t idx = (size_t)blockIdx.x * blockDim.x + threadIdx.x;
  if (idx >= (size_t)Nn * 256) return;
  int i = (int)(idx >> 8), f = (int)(idx & 255);
  int b = batch[i];
  float v = z[idx];
  atomicAdd(&zg[b * 512 + f], v);
  atomicMaxF(&zg[b * 512 + 256 + f], v);
}
__global__ void k_pool_fix(float* __restrict__ zg, const float* __restrict__ cnt) {
  int idx = blockIdx.x * blockDim.x + threadIdx.x;
  if (idx >= 64 * 256) return;
  int b = idx >> 8, f = idx & 255;
  float c = cnt[b];
  zg[b * 512 + f] /= fmaxf(c, 1.f);
  if (c <= 0.f) zg[b * 512 + 256 + f] = 0.f;
}

// ------------------------------------------------------------------ perturbation
__global__ void k_stats(const float* __restrict__ p, size_t n, float* __restrict__ out2) {
  size_t i = (size_t)blockIdx.x * blockDim.x + threadIdx.x;
  size_t stride = (size_t)gridDim.x * blockDim.x;
  float s = 0.f, ss = 0.f;
  for (; i < n; i += stride) { float v = p[i]; s += v; ss += v * v; }
  atomicAdd(&out2[0], s);
  atomicAdd(&out2[1], ss);
}
__device__ __forceinline__ unsigned pcg_hash(unsigned v) {
  v = v * 747796405u + 2891336453u;
  unsigned w = ((v >> ((v >> 28) + 4u)) ^ v) * 277803737u;
  return (w >> 22) ^ w;
}
__global__ void k_perturb(const float* __restrict__ in, float* __restrict__ out,
                          const float* __restrict__ st, float n, unsigned seed, size_t cnt) {
  size_t i = (size_t)blockIdx.x * blockDim.x + threadIdx.x;
  if (i >= cnt) return;
  float mean = st[0] / n;
  float var = (st[1] - st[0] * mean) / fmaxf(n - 1.f, 1.f);
  float sigma = fmaxf(sqrtf(fmaxf(var, 0.f)), 1e-6f);
  unsigned h1 = pcg_hash((unsigned)(i * 2u + 1u) + seed * 0x9E3779B9u);
  unsigned h2 = pcg_hash((unsigned)(i * 2u + 2u) + seed * 0x85EBCA6Bu);
  float u1 = (h1 + 1u) * 2.3283064365386963e-10f;  // (0,1]
  float u2 = h2 * 2.3283064365386963e-10f;
  float z = sqrtf(-2.f * logf(u1)) * cosf(6.28318530717959f * u2);
  out[i] = in[i] + 1.0f /*ETA*/ * sigma * z;
}

// ================================================================== host
extern "C" void kernel_launch(void* const* d_in, const int* in_sizes, int n_in,
                              void* d_out, int out_size, void* d_ws, size_t ws_size,
                              hipStream_t stream) {
  (void)n_in; (void)out_size; (void)ws_size;

  const float* x    = (const float*)d_in[0];
  const int*   ei   = (const int*)d_in[1];
  const int*   batch= (const int*)d_in[2];
  const float* W1   = (const float*)d_in[4];
  const float* b1   = (const float*)d_in[5];
  const float* W2   = (const float*)d_in[6];
  const float* b2   = (const float*)d_in[7];
  const float* Wd   = (const float*)d_in[8];
  const float* bd   = (const float*)d_in[9];
  const float* P1w  = (const float*)d_in[10];
  const float* P1b  = (const float*)d_in[11];
  const float* P2w  = (const float*)d_in[12];
  const float* P2b  = (const float*)d_in[13];
  const float* P3w  = (const float*)d_in[14];
  const float* P3b  = (const float*)d_in[15];

  const int IN = 128, HID = 512, OUT = 256, B = 64;
  const int N = in_sizes[0] / IN;
  const int E = in_sizes[1] / 2;
  const int* src = ei;
  const int* dst = ei + E;

  // ---- workspace carve-out
  char* ws = (char*)d_ws;
  auto carve = [&](size_t bytes) -> float* {
    float* p = (float*)ws;
    ws += ((bytes + 255) / 256) * 256;
    return p;
  };
  float* buf0  = carve((size_t)N * HID * 4);
  float* buf1  = carve((size_t)N * HID * 4);
  float* dinv  = carve((size_t)N * 4);
  float* W1h   = carve((size_t)IN * HID * 4);
  float* b1h   = carve((size_t)HID * 4);
  float* W2h   = carve((size_t)HID * OUT * 4);
  float* b2h   = carve((size_t)OUT * 4);
  float* cnt   = carve((size_t)B * 4);
  float* g1buf = carve((size_t)B * 256 * 4);
  float* g2buf = carve((size_t)B * 128 * 4);
  float* stats = carve(8 * 4);

  // ---- output slices (fp32, tuple order)
  float* z_node    = (float*)d_out;
  float* z_g       = z_node + (size_t)N * OUT;
  float* zhat_node = z_g + (size_t)B * 2 * OUT;
  float* zhat_g    = zhat_node + (size_t)N * OUT;
  float* h_g       = zhat_g + (size_t)B * 2 * OUT;
  float* hhat_g    = h_g + (size_t)B * 64;
  float* x_hat     = hhat_g + (size_t)B * 64;

  auto blocks = [](size_t n) { return (unsigned)((n + 255) / 256); };

  // ---- degrees: deg = 1 + scatter(ones, dst); dinv = rsqrt(deg)
  k_fill<<<blocks(N), 256, 0, stream>>>(dinv, 1.0f, (size_t)N);
  k_deg<<<blocks(E), 256, 0, stream>>>(dst, dinv, E);
  k_rsqrt_inplace<<<blocks(N), 256, 0, stream>>>(dinv, N);

  auto gemm = [&](const float* A, const float* Wm, const float* bias, float* C,
                  int M, int K, int Nc, int relu) {
    dim3 g((Nc + 127) / 128, (M + 63) / 64);
    k_gemm<<<g, 256, 0, stream>>>(A, Wm, bias, C, M, K, Nc, relu);
  };

  auto gcn = [&](const float* Xin, const float* Wm, const float* bias,
                 int K, int F, float* hbuf, float* abuf, float* dest, int relu) {
    gemm(Xin, Wm, nullptr, hbuf, N, K, F, 0);
    size_t nf = (size_t)N * F;
    k_fill<<<blocks(nf), 256, 0, stream>>>(abuf, 0.f, nf);
    long long tot = (long long)E * (F / 4);
    k_scatter<<<blocks((size_t)tot), 256, 0, stream>>>(src, dst, dinv, hbuf, abuf, tot, F / 4, F);
    k_finalize<<<blocks(nf), 256, 0, stream>>>(abuf, hbuf, dinv, bias, dest, N, F, relu);
  };

  auto pool = [&](const float* z, float* zg) {
    k_fill<<<1, 64, 0, stream>>>(cnt, 0.f, (size_t)B);
    k_count<<<blocks(N), 256, 0, stream>>>(batch, cnt, N);
    k_pool_init<<<blocks(B * 256), 256, 0, stream>>>(zg);
    k_pool<<<blocks((size_t)N * 256), 256, 0, stream>>>(z, batch, zg, N);
    k_pool_fix<<<blocks(B * 256), 256, 0, stream>>>(zg, cnt);
  };

  auto proj = [&](const float* zg, float* out) {
    gemm(zg,    P1w, P1b, g1buf, B, 2 * OUT, 256, 1);
    gemm(g1buf, P2w, P2b, g2buf, B, 256, 128, 1);
    gemm(g2buf, P3w, P3b, out,   B, 128, 64, 0);
  };

  // ---- clean branch
  gcn(x,    W1, b1, IN,  HID, buf0, buf1, buf1,   1);  // z1 (ReLU) in buf1
  gcn(buf1, W2, b2, HID, OUT, buf0, buf1, z_node, 0);  // z_node
  pool(z_node, z_g);
  gcn(z_node, Wd, bd, OUT, IN, buf0, buf1, x_hat, 0);  // decoder

  // ---- perturbed weights (SimGRACE)
  k_fill<<<1, 32, 0, stream>>>(stats, 0.f, 8);
  k_stats<<<64, 256, 0, stream>>>(W1, (size_t)IN * HID,  stats + 0);
  k_stats<<<64, 256, 0, stream>>>(b1, (size_t)HID,       stats + 2);
  k_stats<<<64, 256, 0, stream>>>(W2, (size_t)HID * OUT, stats + 4);
  k_stats<<<64, 256, 0, stream>>>(b2, (size_t)OUT,       stats + 6);
  k_perturb<<<blocks((size_t)IN * HID),  256, 0, stream>>>(W1, W1h, stats + 0, (float)(IN * HID),  1u, (size_t)IN * HID);
  k_perturb<<<blocks((size_t)HID),       256, 0, stream>>>(b1, b1h, stats + 2, (float)HID,         2u, (size_t)HID);
  k_perturb<<<blocks((size_t)HID * OUT), 256, 0, stream>>>(W2, W2h, stats + 4, (float)(HID * OUT), 3u, (size_t)HID * OUT);
  k_perturb<<<blocks((size_t)OUT),       256, 0, stream>>>(b2, b2h, stats + 6, (float)OUT,         4u, (size_t)OUT);

  // ---- perturbed branch
  gcn(x,    W1h, b1h, IN,  HID, buf0, buf1, buf1,      1);
  gcn(buf1, W2h, b2h, HID, OUT, buf0, buf1, zhat_node, 0);
  pool(zhat_node, zhat_g);

  // ---- projection heads
  proj(z_g,    h_g);
  proj(zhat_g, hhat_g);
}
